// SpikeLinear_11587821765075
// MI455X (gfx1250) — matrix operand order repaired and checked by reading, not compiled, not run on hardware
//
#include <hip/hip_runtime.h>

// ---------------------------------------------------------------------------
// SpikeLinear fused GEMM + LIF scan for MI455X (gfx1250, wave32, WMMA + TDM)
//
//   pot[n,t,o] = sum_k spike[n,t,k]*W[o,k] + bias[o]/T ;  mem += pot - spk ;
//   spk = (mem >= 1)
//
//  * t-pair register blocking: two f32 accumulators per wave share every B
//    (weight) fragment -> 3 ds_load_b128 per WMMA.
//  * Spike staging via Tensor Data Mover: one tensor_load_to_lds per K-chunk
//    ([t=2][n=32][k=128] f16 tile), TDM pad_enable (+4 dwords per 64 dwords)
//    creates the 272 B padded rows (stride = 4 dwords mod 64 banks ->
//    conflict-free b128 fragment reads). Triple-buffered, issue distance 2.
//  * v4 scalar-path cleanup: rotating buffer offsets (no % 3), unguarded
//    constant s_wait_tensorcnt 1 (non-issuing waves have cnt==0), trailing
//    dummy TDM so the final real chunk is covered by the same constant wait,
//    loop-invariant descriptor groups hoisted.
//  * Weight slice 64x2048 f16 LDS-resident. Total LDS = 315,408 B / 320 KB.
//
// d_ws: [0, 64 MB) spike_in f16, [64 MB, 72 MB) weight f16.
// ---------------------------------------------------------------------------

typedef __attribute__((ext_vector_type(16))) _Float16 v16h;
typedef __attribute__((ext_vector_type(8)))  float    v8f;
typedef unsigned int u32;
typedef unsigned long long u64;
typedef u32 u32x4 __attribute__((ext_vector_type(4)));
typedef int i32x4 __attribute__((ext_vector_type(4)));
typedef int i32x8 __attribute__((ext_vector_type(8)));

#define N_BATCH 128
#define T_STEPS 128
#define IN_F    2048
#define OUT_F   2048

#define BN 32                   // batch rows per block (2 WMMA M-tiles)
#define BO 64                   // output cols per block (4 WMMA N-tiles)
#define TT 2                    // timesteps per register-blocked pair
#define KC 128                  // K chunk (halves) per TDM transfer
#define NCHUNK_K (IN_F / KC)    // 16
#define NTP (T_STEPS / TT)      // 64
#define TOTAL_CHUNKS (NTP * NCHUNK_K)  // 1024

#define SW_STRIDE 2056                       // weight LDS row stride (halves)
#define SW_BYTES (64 * SW_STRIDE * 2)        // 263,168
#define SA_ROW_BYTES ((KC + 8) * 2)          // 272 (TDM-padded spike row)
#define BUF_BYTES (TT * BN * SA_ROW_BYTES)   // 17,408 per chunk buffer
#define SPIKE_OFF SW_BYTES
#define BUF_LAST  (SPIKE_OFF + 2 * BUF_BYTES)
#define DUMMY_OFF (SPIKE_OFF + 3 * BUF_BYTES)   // 16 B scratch for dummy TDM
#define SMEM_BYTES (DUMMY_OFF + 16)             // 315,408 <= 320 KB

#define WS_WEIGHT_OFF ((size_t)N_BATCH * T_STEPS * IN_F * 2)

// ---- prep: f32 -> f16 (8 elems/thread) ----
__global__ void cvt_f32_f16_kernel(const float* __restrict__ in,
                                   _Float16* __restrict__ out) {
  size_t i = ((size_t)blockIdx.x * blockDim.x + threadIdx.x) * 8;
  float4 a = *(const float4*)(in + i);
  float4 b = *(const float4*)(in + i + 4);
  union { _Float16 h[8]; int4 q; } u;
  u.h[0] = (_Float16)a.x; u.h[1] = (_Float16)a.y;
  u.h[2] = (_Float16)a.z; u.h[3] = (_Float16)a.w;
  u.h[4] = (_Float16)b.x; u.h[5] = (_Float16)b.y;
  u.h[6] = (_Float16)b.z; u.h[7] = (_Float16)b.w;
  *(int4*)(out + i) = u.q;
}

__device__ __forceinline__ u32x4 tdm_g0(u64 gaddr, u32 ldsOff) {
  // count=1 | lds_addr | global_addr[56:0] | type=2 ("image")
  u32x4 g0 = { 1u, ldsOff, (u32)gaddr,
               (u32)((gaddr >> 32) & 0x01FFFFFFull) | (2u << 30) };
  return g0;
}

// ---- fused WMMA GEMM + LIF scan ----
__launch_bounds__(256, 1)
__global__ void snn_wmma_kernel(const _Float16* __restrict__ spk16,
                                const _Float16* __restrict__ wgt16,
                                const float* __restrict__ bias,
                                float* __restrict__ out) {
  extern __shared__ char smem[];

  const unsigned tid     = threadIdx.x;
  const unsigned lane    = tid & 31u;
  const unsigned wv      = tid >> 5;     // 0..7
  const unsigned mw      = wv & 1u;      // M-tile
  const unsigned cw      = wv >> 1;      // N-tile 0..3
  const unsigned laneLow = lane & 15u;
  const unsigned laneHi  = lane >> 4;

  const unsigned o_base = (blockIdx.x & 31u) * BO;
  const unsigned n_base = (blockIdx.x >> 5) * BN;

  // ---- loop-invariant TDM descriptor groups (ISA ch.8) ----
  // 3D tensor: x=k (contig), y=n (stride T*IN), z=t (stride IN), f16 data.
  // pad_enable: +4 dwords (code 3) after every 64 dwords (code 5).
  const i32x8 g1r = { (int)((1u << 16) | (1u << 20) | (5u << 22) | (3u << 25)),
                      (int)((u32)IN_F << 16),       // tensor_dim0 = 2048 (k)
                      (int)((u32)N_BATCH << 16),    // tensor_dim1 = 128  (n)
                      (int)((u32)KC << 16),         // tile_dim0   = 128
                      (int)(32u | (2u << 16)),      // tile_dim1=32, tile_dim2=2
                      (int)(T_STEPS * IN_F),        // dim0 stride = 262144
                      (int)((u32)IN_F << 16),       // dim1 stride = 2048
                      0 };
  const i32x4 g2r = { T_STEPS, 0, 0, 0 };           // tensor_dim2 = 128 (t)
  // dummy: 1D, 8-half (16 B) load into scratch; keeps TENSORcnt retirement
  // in-order behind the last real chunk so the wait immediate stays constant.
  const i32x8 g1d = { (int)(1u << 16),              // data_size=2B, no pad
                      (int)((u32)IN_F << 16),       // tensor_dim0 = 2048
                      0,
                      (int)(8u << 16),              // tile_dim0 = 8
                      0, 0, 0, 0 };
  const i32x4 gz4 = { 0, 0, 0, 0 };
  const i32x8 gz8 = { 0, 0, 0, 0, 0, 0, 0, 0 };

  const u64 gbase = (u64)spk16 + (u64)n_base * T_STEPS * IN_F * 2ull;

  // kick off the TDM pipeline before weight staging so the copies overlap it
  if (wv == 0) {
    __builtin_amdgcn_tensor_load_to_lds(tdm_g0(gbase, SPIKE_OFF),
                                        g1r, g2r, gz4, gz8, 0);
    __builtin_amdgcn_tensor_load_to_lds(tdm_g0(gbase + KC * 2, SPIKE_OFF + BUF_BYTES),
                                        g1r, g2r, gz4, gz8, 0);
  }

  // ---- stage weight slice (64 x 2048 f16) into LDS once ----
  {
    const unsigned r  = tid >> 2;            // 0..63
    const unsigned cs = (tid & 3u) * 512u;
    const _Float16* src = wgt16 + (size_t)(o_base + r) * IN_F + cs;
    char* dst = smem + (size_t)r * (SW_STRIDE * 2) + (size_t)cs * 2;
#pragma unroll 8
    for (unsigned i = 0; i < 512; i += 8)
      *(int4*)(dst + i * 2) = *(const int4*)(src + i);
  }
  __syncthreads();

  const unsigned col = o_base + 16u * cw + laneLow;
  const float bfrac = bias[col] * (1.0f / (float)T_STEPS);

  v8f mem = {};
  v8f spk = {};

  // fragment base byte offsets (ISA 7.12.2 layouts, wave32)
  const unsigned aRowOff = (16u * mw + laneLow) * SA_ROW_BYTES + 16u * laneHi;
  const char* bBase = smem + (size_t)(16u * cw + laneLow) * (SW_STRIDE * 2)
                           + 16u * laneHi;

  size_t obase = ((size_t)(n_base + 16u * mw + 8u * laneHi)) * T_STEPS * OUT_F + col;

  u32 rdOff = SPIKE_OFF;                 // buffer holding chunk c
  u32 wrOff = SPIKE_OFF + 2 * BUF_BYTES; // buffer for chunk c+2
  int c = 0;
#pragma unroll 1
  for (int tp = 0; tp < NTP; ++tp) {
    // accumulator seeds: c0 = mem - Vthr*spk (state), c1 = 0 (pure pot)
    v8f c0, c1;
#pragma unroll
    for (int j = 0; j < 8; ++j) { c0[j] = mem[j] - spk[j]; c1[j] = 0.0f; }

#pragma unroll 1
    for (int kbi = 0; kbi < NCHUNK_K; ++kbi, ++c) {
      // oldest in-flight chunk (c) has landed; non-issuing waves have cnt==0
      __builtin_amdgcn_s_wait_tensorcnt(1);
      __syncthreads();   // publish chunk c; close all reads of buf[(c+2)%3]

      if (wv == 0) {
        const int cc = c + 2;
        if (cc < TOTAL_CHUNKS) {
          const u64 gaddr = gbase + (u64)(cc >> 4) * (TT * IN_F * 2)
                                  + (u64)(cc & (NCHUNK_K - 1)) * (KC * 2);
          __builtin_amdgcn_tensor_load_to_lds(tdm_g0(gaddr, wrOff),
                                              g1r, g2r, gz4, gz8, 0);
        } else if (cc == TOTAL_CHUNKS) {
          __builtin_amdgcn_tensor_load_to_lds(tdm_g0(gbase, DUMMY_OFF),
                                              g1d, gz4, gz4, gz8, 0);
        }
      }
      wrOff = (wrOff == BUF_LAST) ? (u32)SPIKE_OFF : wrOff + BUF_BYTES;

      const char* buf = smem + rdOff;
      const char* a0  = buf + aRowOff;                 // t = 2*tp
      const char* a1  = a0 + BN * SA_ROW_BYTES;        // t = 2*tp+1
      const int kb = kbi * KC;
      rdOff = (rdOff == BUF_LAST) ? (u32)SPIKE_OFF : rdOff + BUF_BYTES;

#pragma unroll
      for (int kk = 0; kk < KC; kk += 32) {
        union { v16h v; int4 q[2]; } A0, A1, B;
        A0.q[0] = *(const int4*)(a0 + kk * 2);
        A0.q[1] = *(const int4*)(a0 + kk * 2 + 32);
        A1.q[0] = *(const int4*)(a1 + kk * 2);
        A1.q[1] = *(const int4*)(a1 + kk * 2 + 32);
        B.q[0]  = *(const int4*)(bBase + (kb + kk) * 2);
        B.q[1]  = *(const int4*)(bBase + (kb + kk) * 2 + 32);
        c0 = __builtin_amdgcn_wmma_f32_16x16x32_f16(
            false, A0.v, false, B.v, (short)0, c0, false, false);
        c1 = __builtin_amdgcn_wmma_f32_16x16x32_f16(
            false, A1.v, false, B.v, (short)0, c1, false, false);
      }
    }

    // LIF update for the t-pair + spike stores (8 rows per lane)
#pragma unroll
    for (int j = 0; j < 8; ++j) {
      float m1 = c0[j] + bfrac;                 // mem after t = 2*tp
      float s1 = (m1 >= 1.0f) ? 1.0f : 0.0f;
      float m2 = m1 - s1 + c1[j] + bfrac;       // mem after t = 2*tp+1
      float s2 = (m2 >= 1.0f) ? 1.0f : 0.0f;
      out[obase + (size_t)j * (T_STEPS * OUT_F)]         = s1;
      out[obase + (size_t)j * (T_STEPS * OUT_F) + OUT_F] = s2;
      mem[j] = m2;
      spk[j] = s2;
    }
    obase += 2 * OUT_F;
  }
}

extern "C" void kernel_launch(void* const* d_in, const int* in_sizes, int n_in,
                              void* d_out, int out_size, void* d_ws, size_t ws_size,
                              hipStream_t stream) {
  const float* spike_f32  = (const float*)d_in[0];   // [128,128,2048]
  const float* weight_f32 = (const float*)d_in[1];   // [2048,2048]
  const float* bias       = (const float*)d_in[2];   // [2048]
  float* out = (float*)d_out;

  _Float16* wsSpike  = (_Float16*)d_ws;
  _Float16* wsWeight = (_Float16*)((char*)d_ws + WS_WEIGHT_OFF);

  const int nSpike  = N_BATCH * T_STEPS * IN_F;
  const int nWeight = OUT_F * IN_F;
  cvt_f32_f16_kernel<<<nSpike  / (256 * 8), 256, 0, stream>>>(spike_f32,  wsSpike);
  cvt_f32_f16_kernel<<<nWeight / (256 * 8), 256, 0, stream>>>(weight_f32, wsWeight);

  (void)hipFuncSetAttribute((const void*)snn_wmma_kernel,
                            hipFuncAttributeMaxDynamicSharedMemorySize,
                            SMEM_BYTES);

  dim3 grid((N_BATCH / BN) * (OUT_F / BO));  // 128 blocks
  dim3 block(256);                           // 8 waves
  snn_wmma_kernel<<<grid, block, SMEM_BYTES, stream>>>(wsSpike, wsWeight, bias, out);
}
